// MultiheadAttentionMechanism_14207751815518
// MI455X (gfx1250) — matrix operand order repaired and checked by reading, not compiled, not run on hardware
//
#include <hip/hip_runtime.h>
#include <hip/hip_bf16.h>

// ---------------------------------------------------------------------------
// MI455X (gfx1250) multi-head attention, fp32 I/O, f16 WMMA compute,
// async global->LDS staging (GLOBAL_LOAD_ASYNC_TO_LDS_B128 / ASYNCcnt).
// ---------------------------------------------------------------------------

typedef __attribute__((ext_vector_type(16))) _Float16 v16h;
typedef __attribute__((ext_vector_type(8)))  float    v8f;
typedef __attribute__((ext_vector_type(4)))  int      i32x4;

#define BDIM    4
#define LDIM    1024
#define DMODEL  1024
#define NHEAD   16
#define DHEAD   64
#define INV_SCALE 0.125f                       // 1/sqrt(64)
#define NEG_INF  (-3.4028234663852886e+38f)    // float32 min (matches reference)

// ---- async global->LDS (GLOBAL_LOAD_ASYNC_TO_LDS_B128, ASYNCcnt) ----------
#if defined(__has_builtin)
#if __has_builtin(__builtin_amdgcn_global_load_async_to_lds_b128)
#define USE_ASYNC_LDS 1
#endif
#endif

#ifdef USE_ASYNC_LDS
typedef __attribute__((address_space(1))) i32x4* gas_b128_t;   // global src
typedef __attribute__((address_space(3))) i32x4* las_b128_t;   // LDS dst
#endif

__device__ __forceinline__ void cp_g2l_f32x4(const float* g, float* l) {
#ifdef USE_ASYNC_LDS
    __builtin_amdgcn_global_load_async_to_lds_b128((gas_b128_t)g, (las_b128_t)l,
                                                   0, 0);
#else
    *(float4*)l = *(const float4*)g;
#endif
}

__device__ __forceinline__ void async_join() {
#ifdef USE_ASYNC_LDS
#if __has_builtin(__builtin_amdgcn_s_wait_asynccnt)
    __builtin_amdgcn_s_wait_asynccnt(0);
#else
    asm volatile("s_wait_asynccnt 0x0" ::: "memory");
#endif
#endif
}

// --- ISA 7.12.2 fragment index maps (16-bit data, wave32) -------------------
// A-matrix 16x32 (MxK): lane%16 = M; element e -> K per half-wave interleave.
__device__ __forceinline__ int a_frag_k(int lane, int e) {
    int g = e >> 1, p = e & 1;
    int lo = (lane & 16) ? 8 : 0;
    int base = (g < 4) ? (2 * g + lo) : (2 * g + 8 + lo);
    return base + p;
}
// B-matrix 32x16 (KxN): lane%16 = N; element e -> K, same half-wave interleave as A.
__device__ __forceinline__ int b_frag_k(int lane, int e) {
    int lo = (lane & 16) ? 8 : 0;
    return (e < 8) ? (e + lo) : (e + 8 + lo);
}
// C/D 16x16 f32: element r -> M = r + 8*(lane/16), N = lane%16.

// ---------------------------------------------------------------------------
// Kernel 1/4: C[M,N] = A[M,K] @ W[N,K]^T + bias[N]   (fp32 in/out, f16 WMMA)
// block 256 (8 waves), tile 128(M) x 64(N), K-step 32; fp32 LDS tiles staged
// asynchronously, f32->f16 conversion folded into fragment build.
// ---------------------------------------------------------------------------
__global__ __launch_bounds__(256)
void gemm_xwt_bias(const float* __restrict__ A, const float* __restrict__ W,
                   const float* __restrict__ bias, float* __restrict__ C,
                   int M, int N, int K) {
    __shared__ float As[128][36];   // fp32 tile, +4 pad (row = 144B, 16B aligned)
    __shared__ float Bs[64][36];

    const int tid  = threadIdx.x;
    const int wave = tid >> 5, lane = tid & 31;
    const int wm = (wave >> 1) * 32;          // 0,32,64,96
    const int wn = (wave & 1) * 32;           // 0,32
    const int bm = blockIdx.y * 128;
    const int bn = blockIdx.x * 64;

    v8f acc[2][2] = {};

    for (int k0 = 0; k0 < K; k0 += 32) {
        if (k0 + 32 < K)
            __builtin_prefetch(&A[(size_t)(bm + (tid >> 1)) * K + k0 + 32], 0, 0);

        // A: 128x32 f32 = 1024 16B-chunks -> 4/thread ; B: 64x32 -> 2/thread
#pragma unroll
        for (int t = 0; t < 4; ++t) {
            int idx = tid + t * 256;
            int r = idx >> 3, c4 = (idx & 7) * 4;
            cp_g2l_f32x4(&A[(size_t)(bm + r) * K + k0 + c4], &As[r][c4]);
        }
#pragma unroll
        for (int t = 0; t < 2; ++t) {
            int idx = tid + t * 256;
            int r = idx >> 3, c4 = (idx & 7) * 4;
            cp_g2l_f32x4(&W[(size_t)(bn + r) * K + k0 + c4], &Bs[r][c4]);
        }
        async_join();
        __syncthreads();

        v16h af[2], bf[2];
#pragma unroll
        for (int mi = 0; mi < 2; ++mi)
#pragma unroll
            for (int e = 0; e < 16; ++e)
                af[mi][e] = (_Float16)As[wm + mi * 16 + (lane & 15)][a_frag_k(lane, e)];
#pragma unroll
        for (int ni = 0; ni < 2; ++ni)
#pragma unroll
            for (int e = 0; e < 16; ++e)
                bf[ni][e] = (_Float16)Bs[wn + ni * 16 + (lane & 15)][b_frag_k(lane, e)];

#pragma unroll
        for (int mi = 0; mi < 2; ++mi)
#pragma unroll
            for (int ni = 0; ni < 2; ++ni)
                acc[mi][ni] = __builtin_amdgcn_wmma_f32_16x16x32_f16(
                    false, af[mi], false, bf[ni], (short)0, acc[mi][ni], false, false);
        __syncthreads();
    }

#pragma unroll
    for (int mi = 0; mi < 2; ++mi)
#pragma unroll
        for (int ni = 0; ni < 2; ++ni)
#pragma unroll
            for (int r = 0; r < 8; ++r) {
                int m = bm + wm + mi * 16 + r + 8 * (lane >> 4);
                int n = bn + wn + ni * 16 + (lane & 15);
                C[(size_t)m * N + n] = acc[mi][ni][r] + bias[n];
            }
}

// ---------------------------------------------------------------------------
// Kernel 2: scores = mask(QK^T/8), softmax over keys, write aw (B,H,L,L).
// One block per (16-row i-tile, h, b). Full 16x1024 fp32 score stripe in LDS
// (64KB) + fp32 K chunk (34KB) + fp32 Q tile (4.3KB): ~103KB of the 320KB WGP
// LDS — the scores never round-trip through HBM.
// ---------------------------------------------------------------------------
__global__ __launch_bounds__(256)
void attn_scores_softmax(const float* __restrict__ q, const float* __restrict__ k,
                         const int* __restrict__ mask, float* __restrict__ aw) {
    extern __shared__ unsigned char smem[];
    float* Sc = (float*)smem;                          // [16][1024]
    float* Ks = Sc + 16 * LDIM;                        // [128][68]
    float* Qs = Ks + 128 * 68;                         // [16][68]

    const int tid  = threadIdx.x;
    const int wave = tid >> 5, lane = tid & 31;
    const int i0 = blockIdx.x * 16;
    const int h  = blockIdx.y;
    const int b  = blockIdx.z;

    // stage Q tile 16 x 64 f32 (256 chunks of 16B -> 1/thread)
    {
        int r = tid >> 4, c4 = (tid & 15) * 4;
        cp_g2l_f32x4(&q[(size_t)(b * LDIM + i0 + r) * DMODEL + h * DHEAD + c4],
                     &Qs[r * 68 + c4]);
    }
    async_join();
    __syncthreads();

    v16h qf[2];
#pragma unroll
    for (int ks = 0; ks < 2; ++ks)
#pragma unroll
        for (int e = 0; e < 16; ++e)
            qf[ks][e] = (_Float16)Qs[(lane & 15) * 68 + ks * 32 + a_frag_k(lane, e)];

    for (int chunk = 0; chunk < 8; ++chunk) {
        const int j0 = chunk * 128;
        __syncthreads();                                  // Ks reusable
        // K chunk: 128x64 f32 = 2048 16B-chunks -> 8/thread
#pragma unroll
        for (int t = 0; t < 8; ++t) {
            int idx = tid + t * 256;
            int r = idx >> 4, c4 = (idx & 15) * 4;
            cp_g2l_f32x4(&k[(size_t)(b * LDIM + j0 + r) * DMODEL + h * DHEAD + c4],
                         &Ks[r * 68 + c4]);
        }
        async_join();
        __syncthreads();

        // wave w computes score sub-tile j = j0 + 16*w .. +15
        v8f acc = {};
#pragma unroll
        for (int ks = 0; ks < 2; ++ks) {
            v16h bf;
#pragma unroll
            for (int e = 0; e < 16; ++e)
                bf[e] = (_Float16)Ks[(wave * 16 + (lane & 15)) * 68 + ks * 32 +
                                     b_frag_k(lane, e)];
            acc = __builtin_amdgcn_wmma_f32_16x16x32_f16(
                false, qf[ks], false, bf, (short)0, acc, false, false);
        }
        // scale + mask at fragment store
#pragma unroll
        for (int r = 0; r < 8; ++r) {
            int m  = r + 8 * (lane >> 4);
            int jj = wave * 16 + (lane & 15);
            int mk = mask[(size_t)(b * LDIM + i0 + m) * LDIM + j0 + jj];
            Sc[m * LDIM + j0 + jj] = mk ? acc[r] * INV_SCALE : NEG_INF;
        }
    }
    __syncthreads();

    // softmax: 16 lanes per row (shfl_xor width-16 reductions)
    const int row = tid >> 4, sl = tid & 15;
    float mx = NEG_INF;
    for (int c = sl; c < LDIM; c += 16) mx = fmaxf(mx, Sc[row * LDIM + c]);
#pragma unroll
    for (int o = 8; o >= 1; o >>= 1) mx = fmaxf(mx, __shfl_xor(mx, o, 16));
    float sum = 0.f;
    for (int c = sl; c < LDIM; c += 16) {
        float p = __expf(Sc[row * LDIM + c] - mx);
        Sc[row * LDIM + c] = p;
        sum += p;
    }
#pragma unroll
    for (int o = 8; o >= 1; o >>= 1) sum += __shfl_xor(sum, o, 16);
    const float inv = 1.f / sum;

    float* awp = aw + ((size_t)(b * NHEAD + h) * LDIM + (i0 + row)) * LDIM;
    for (int c = sl; c < LDIM; c += 16) awp[c] = Sc[row * LDIM + c] * inv;
}

// ---------------------------------------------------------------------------
// Kernel 3: cv[b,i,h*64+d] = sum_j aw[b,h,i,j] * v[b,j,h*64+d]
// block 256 (8 waves), tile 128(i) x 64(d), K-step 32 over j.
// ---------------------------------------------------------------------------
__global__ __launch_bounds__(256)
void attn_av(const float* __restrict__ aw, const float* __restrict__ v,
             float* __restrict__ cv) {
    __shared__ float As[128][36];   // aw tile (i x j), fp32
    __shared__ float Vs[32][68];    // v  tile (j x d), fp32

    const int tid  = threadIdx.x;
    const int wave = tid >> 5, lane = tid & 31;
    const int i0 = blockIdx.x * 128;
    const int h  = blockIdx.y;
    const int b  = blockIdx.z;
    const float* awb = aw + (size_t)(b * NHEAD + h) * LDIM * LDIM;

    v8f acc[4] = {};
    for (int j0 = 0; j0 < LDIM; j0 += 32) {
        // aw: 128x32 -> 1024 chunks (4/thread); v: 32x64 -> 512 chunks (2/thread)
#pragma unroll
        for (int t = 0; t < 4; ++t) {
            int idx = tid + t * 256;
            int r = idx >> 3, c4 = (idx & 7) * 4;
            cp_g2l_f32x4(&awb[(size_t)(i0 + r) * LDIM + j0 + c4], &As[r][c4]);
        }
#pragma unroll
        for (int t = 0; t < 2; ++t) {
            int idx = tid + t * 256;
            int r = idx >> 4, c4 = (idx & 15) * 4;
            cp_g2l_f32x4(&v[(size_t)(b * LDIM + j0 + r) * DMODEL + h * DHEAD + c4],
                         &Vs[r][c4]);
        }
        async_join();
        __syncthreads();

        v16h af;
#pragma unroll
        for (int e = 0; e < 16; ++e)
            af[e] = (_Float16)As[wave * 16 + (lane & 15)][a_frag_k(lane, e)];
#pragma unroll
        for (int ni = 0; ni < 4; ++ni) {
            v16h bf;
#pragma unroll
            for (int e = 0; e < 16; ++e)
                bf[e] = (_Float16)Vs[b_frag_k(lane, e)][ni * 16 + (lane & 15)];
            acc[ni] = __builtin_amdgcn_wmma_f32_16x16x32_f16(
                false, af, false, bf, (short)0, acc[ni], false, false);
        }
        __syncthreads();
    }

#pragma unroll
    for (int ni = 0; ni < 4; ++ni)
#pragma unroll
        for (int r = 0; r < 8; ++r) {
            int m = i0 + wave * 16 + r + 8 * (lane >> 4);
            int d = ni * 16 + (lane & 15);
            cv[(size_t)(b * LDIM + m) * DMODEL + h * DHEAD + d] = acc[ni][r];
        }
}

// ---------------------------------------------------------------------------
extern "C" void kernel_launch(void* const* d_in, const int* in_sizes, int n_in,
                              void* d_out, int out_size, void* d_ws, size_t ws_size,
                              hipStream_t stream) {
    (void)in_sizes; (void)n_in; (void)out_size; (void)ws_size;

    const float* key   = (const float*)d_in[0];
    const float* value = (const float*)d_in[1];
    const float* query = (const float*)d_in[2];
    const int*   mask  = (const int*)d_in[3];
    const float* Wk = (const float*)d_in[4];  const float* bk = (const float*)d_in[5];
    const float* Wv = (const float*)d_in[6];  const float* bv = (const float*)d_in[7];
    const float* Wq = (const float*)d_in[8];  const float* bq = (const float*)d_in[9];
    const float* Wo = (const float*)d_in[10]; const float* bo = (const float*)d_in[11];

    float* out = (float*)d_out;                                 // (B,L,1024)
    float* aw  = out + (size_t)BDIM * LDIM * DMODEL;            // (B,H,L,L)

    const size_t per = (size_t)BDIM * LDIM * DMODEL;            // 4M floats
    float* qws  = (float*)d_ws;
    float* kws  = qws + per;
    float* vws  = kws + per;
    float* cvws = vws + per;

    const dim3 blk(256);
    const dim3 gGemm(DMODEL / 64, (BDIM * LDIM) / 128);         // 16 x 32

    // 1) projections (q, k, v)
    gemm_xwt_bias<<<gGemm, blk, 0, stream>>>(query, Wq, bq, qws,
                                             BDIM * LDIM, DMODEL, DMODEL);
    gemm_xwt_bias<<<gGemm, blk, 0, stream>>>(key,   Wk, bk, kws,
                                             BDIM * LDIM, DMODEL, DMODEL);
    gemm_xwt_bias<<<gGemm, blk, 0, stream>>>(value, Wv, bv, vws,
                                             BDIM * LDIM, DMODEL, DMODEL);

    // 2) scores + mask + softmax -> aw   (dynamic LDS: 64KB Sc + 34KB K + 4.3KB Q)
    const size_t smem2 = (size_t)(16 * LDIM + 128 * 68 + 16 * 68) * sizeof(float);
    attn_scores_softmax<<<dim3(LDIM / 16, NHEAD, BDIM), blk, smem2, stream>>>(
        qws, kws, mask, aw);

    // 3) cv = aw @ V
    attn_av<<<dim3(LDIM / 128, NHEAD, BDIM), blk, 0, stream>>>(aw, vws, cvws);

    // 4) output projection
    gemm_xwt_bias<<<gGemm, blk, 0, stream>>>(cvws, Wo, bo, out,
                                             BDIM * LDIM, DMODEL, DMODEL);
}
